// TransformerFlatten_19396072308744
// MI455X (gfx1250) — compile-verified, matrix operand order checked by reading, not run on hardware
//
#include <hip/hip_runtime.h>
#include <hip/hip_bf16.h>

// dims: L=S=1024, N=4, E=1024, H=16, HD=64, NH=64 batches
#define L 1024
#define S 1024
#define NB 4
#define E 1024
#define H 16
#define HD 64

typedef __bf16 bf16_t;
typedef __attribute__((ext_vector_type(4)))  __bf16 v4bf;
typedef __attribute__((ext_vector_type(8)))  __bf16 v8bf;
typedef __attribute__((ext_vector_type(16))) __bf16 v16bf;
typedef __attribute__((ext_vector_type(8)))  float  v8f;
// exact parameter type of the async-LDS builtin per hipcc diagnostic:
typedef int v4i_vs __attribute__((vector_size(16)));

// CDNA5 async global->LDS copy path (ASYNCcnt-tracked), if the toolchain
// exposes it. Falls back to a regular b128 copy otherwise.
#if defined(__gfx1250__) && __has_builtin(__builtin_amdgcn_global_load_async_to_lds_b128)
#define HAVE_ASYNC_LDS 1
#else
#define HAVE_ASYNC_LDS 0
#endif

__device__ __forceinline__ void wait_async_zero() {
#if defined(__gfx1250__)
#if __has_builtin(__builtin_amdgcn_s_wait_asynccnt)
  __builtin_amdgcn_s_wait_asynccnt(0);
#else
  asm volatile("s_wait_asynccnt 0x0" ::: "memory");
#endif
#endif
}

// ---------------------------------------------------------------------------
// WMMA helpers (CDNA5 wave32, V_WMMA_F32_16X16X32_BF16)
// ---------------------------------------------------------------------------
__device__ __forceinline__ v8f wmma_bf16(v16bf a, v16bf b, v8f c) {
  return __builtin_amdgcn_wmma_f32_16x16x32_bf16(false, a, false, b,
                                                 (short)0, c, false, false);
}

// 16x32 (MxK) bf16 A-fragment (or 32x16 KxN B-fragment, symmetric layout)
// from a row-major [row][k] buffer, ISA 7.12.2 lane layout:
//   lane 0-15  : row = lane,    K = 0..7  and 16..23
//   lane 16-31 : row = lane-16, K = 8..15 and 24..31
__device__ __forceinline__ v16bf load_frag(const bf16_t* __restrict__ p,
                                           int stride, int lane) {
  const int row = lane & 15;
  const int kb  = (lane >> 4) << 3;   // 0 or 8
  const bf16_t* r0 = p + row * stride + kb;
  v16bf f;
#pragma unroll
  for (int i = 0; i < 8; ++i) {
    f[i]     = r0[i];                 // K = kb .. kb+7      (16B chunk)
    f[i + 8] = r0[i + 16];            // K = kb+16 .. kb+23  (16B chunk)
  }
  return f;
}

__device__ __forceinline__ v8bf cvt8(float4 a, float4 b) {
  v8bf o;
  o[0] = (__bf16)a.x; o[1] = (__bf16)a.y; o[2] = (__bf16)a.z; o[3] = (__bf16)a.w;
  o[4] = (__bf16)b.x; o[5] = (__bf16)b.y; o[6] = (__bf16)b.z; o[7] = (__bf16)b.w;
  return o;
}

// Stage one 64-wide row chunk (16 f32 -> 16 bf16) per thread: 4x b128 global
// loads, 2x b128 LDS stores. tid covers a 64x64 slab with 256 threads.
__device__ __forceinline__ void stage_f32(bf16_t (*__restrict__ dst)[64],
                                          const float* __restrict__ src,
                                          int row_base, int k0, int tid,
                                          bool pf) {
  const int r  = tid >> 2;
  const int c0 = (tid & 3) << 4;
  const float* p = src + (size_t)(row_base + r) * E + k0 + c0;
  const float4 x0 = ((const float4*)p)[0];
  const float4 x1 = ((const float4*)p)[1];
  const float4 x2 = ((const float4*)p)[2];
  const float4 x3 = ((const float4*)p)[3];
  *(v8bf*)&dst[r][c0]     = cvt8(x0, x1);
  *(v8bf*)&dst[r][c0 + 8] = cvt8(x2, x3);
  if (pf) __builtin_prefetch(p + 64, 0, 1);   // next K slab, same row
}

// ---------------------------------------------------------------------------
// Kernel 1: fused QKV projection.
// C[m, j] = X[m, :] . W_in[j, :] + b_in[j]   (m = l*N+n, j in [0,3E))
// q scaled by HD^-0.5; scattered bf16 into q[b][l][d], k[b][s][d], v_t[b][d][s]
// ---------------------------------------------------------------------------
__global__ __launch_bounds__(256) void qkv_proj_kernel(
    const float* __restrict__ query, const float* __restrict__ key,
    const float* __restrict__ value, const float* __restrict__ w_in,
    const float* __restrict__ b_in, bf16_t* __restrict__ q_bf,
    bf16_t* __restrict__ k_bf, bf16_t* __restrict__ v_t) {
  __shared__ bf16_t As[64][64];   // 8 KB
  __shared__ bf16_t Bs[64][64];   // 8 KB

  const int tid  = threadIdx.x;
  const int lane = tid & 31;
  const int wave = tid >> 5;
  const int m0 = blockIdx.x * 64;          // output row base
  const int cb = blockIdx.y * 64;          // output col base (within 3E)
  const int sel = cb >> 10;                // 0=q, 1=k, 2=v source
  const float* __restrict__ X = (sel == 0) ? query : (sel == 1 ? key : value);

  const int tm0 = wave >> 2;
  const int tm1 = tm0 + 2;
  const int tn  = wave & 3;

  v8f acc0 = {};
  v8f acc1 = {};

  for (int kk = 0; kk < E; kk += 64) {
    const bool pf = (kk + 64) < E;
    stage_f32(As, X, m0, kk, tid, pf);
    stage_f32(Bs, w_in, cb, kk, tid, pf);
    __syncthreads();
#pragma unroll
    for (int ks = 0; ks < 64; ks += 32) {
      v16bf a0 = load_frag(&As[tm0 * 16][ks], 64, lane);
      v16bf a1 = load_frag(&As[tm1 * 16][ks], 64, lane);
      v16bf b0 = load_frag(&Bs[tn * 16][ks], 64, lane);
      acc0 = wmma_bf16(a0, b0, acc0);
      acc1 = wmma_bf16(a1, b0, acc1);
    }
    __syncthreads();
  }

  const int col  = lane & 15;
  const int row0 = (lane >> 4) << 3;
#pragma unroll
  for (int t = 0; t < 2; ++t) {
    const v8f acc = t ? acc1 : acc0;
    const int tm = t ? tm1 : tm0;
#pragma unroll
    for (int r = 0; r < 8; ++r) {
      const int grow = m0 + tm * 16 + row0 + r;   // m = l*N + n
      const int gcol = cb + tn * 16 + col;        // j in [0, 3E)
      float v = acc[r] + b_in[gcol];
      const int ls = grow >> 2;                   // l (or s)
      const int n  = grow & 3;
      const int j  = gcol & (E - 1);
      const int h = j >> 6, d = j & 63;
      const int b = n * H + h;
      if (gcol < E) {
        q_bf[(b * L + ls) * HD + d] = (bf16_t)(v * 0.125f);
      } else if (gcol < 2 * E) {
        k_bf[(b * S + ls) * HD + d] = (bf16_t)v;
      } else {
        v_t[(b * HD + d) * S + ls] = (bf16_t)v;
      }
    }
  }
}

// ---------------------------------------------------------------------------
// Kernel 2: attention. Block = (16-row l-tile, batch n), 4 waves, loops all
// 16 heads so head-summed attn/sig outputs need no atomics.
// ---------------------------------------------------------------------------
__global__ __launch_bounds__(128) void attention_kernel(
    const bf16_t* __restrict__ q_bf, const bf16_t* __restrict__ k_bf,
    const bf16_t* __restrict__ v_t, bf16_t* __restrict__ out_h,
    float* __restrict__ attn_out, float* __restrict__ sig_out) {
  __shared__ float  sc[16][S];      // 64 KB score strip
  __shared__ bf16_t pb[16][S];      // 32 KB bf16 probs
  __shared__ float  red[16][8];

  const int tid  = threadIdx.x;
  const int lane = tid & 31;
  const int wave = tid >> 5;
  const int l0 = blockIdx.x * 16;
  const int n  = blockIdx.y;
  const int col  = lane & 15;
  const int row0 = (lane >> 4) << 3;
  const int srow = tid >> 3;        // softmax: 8 threads per row
  const int seg  = tid & 7;
  const int cbeg = seg * 128, cend = cbeg + 128;

  for (int h = 0; h < H; ++h) {
    const int b = n * H + h;

    // ---- scores: q (16xHD) @ k^T, K = HD = 64 -> 2 WMMAs per 16x16 tile ----
    const bf16_t* qbase = q_bf + (b * L + l0) * HD;
    v16bf qa0 = load_frag(qbase + 0, HD, lane);
    v16bf qa1 = load_frag(qbase + 32, HD, lane);
#pragma unroll 2
    for (int st = wave; st < S / 16; st += 4) {
      const int s = st * 16;
      const bf16_t* kbase = k_bf + (b * S + s) * HD;
      v16bf kb0 = load_frag(kbase + 0, HD, lane);
      v16bf kb1 = load_frag(kbase + 32, HD, lane);
      v8f acc = {};
      acc = wmma_bf16(qa0, kb0, acc);
      acc = wmma_bf16(qa1, kb1, acc);
#pragma unroll
      for (int r = 0; r < 8; ++r) sc[row0 + r][s + col] = acc[r];
    }
    __syncthreads();

    // ---- row max (vectorized) ----
    float mx = -3.0e38f;
    for (int c = cbeg; c < cend; c += 4) {
      const float4 x = *(const float4*)&sc[srow][c];
      mx = fmaxf(mx, fmaxf(fmaxf(x.x, x.y), fmaxf(x.z, x.w)));
    }
    red[srow][seg] = mx;
    __syncthreads();
    float rmax = red[srow][0];
#pragma unroll
    for (int i = 1; i < 8; ++i) rmax = fmaxf(rmax, red[srow][i]);
    __syncthreads();

    // ---- exp + sigmoid accumulation; sigmoid(x) = e / (e + exp(-rmax)) ----
    const float er = __expf(-rmax);
    float sum = 0.f;
    float* sgrow = sig_out + ((size_t)(n * L + l0 + srow)) * S;
    for (int c = cbeg; c < cend; c += 4) {
      const float4 x = *(const float4*)&sc[srow][c];
      float4 e, sg, pv;
      e.x = __expf(x.x - rmax); e.y = __expf(x.y - rmax);
      e.z = __expf(x.z - rmax); e.w = __expf(x.w - rmax);
      sg.x = e.x / (e.x + er);  sg.y = e.y / (e.y + er);
      sg.z = e.z / (e.z + er);  sg.w = e.w / (e.w + er);
      if (h == 0) { pv.x = pv.y = pv.z = pv.w = 0.f; }
      else        { pv = *(const float4*)&sgrow[c]; }
      pv.x += sg.x * (1.0f / H); pv.y += sg.y * (1.0f / H);
      pv.z += sg.z * (1.0f / H); pv.w += sg.w * (1.0f / H);
      *(float4*)&sgrow[c] = pv;
      *(float4*)&sc[srow][c] = e;
      sum += (e.x + e.y) + (e.z + e.w);
    }
    red[srow][seg] = sum;
    __syncthreads();
    float rsum = 0.f;
#pragma unroll
    for (int i = 0; i < 8; ++i) rsum += red[srow][i];
    const float inv = 1.0f / rsum;

    // ---- normalize -> bf16 probs + attn-weight accumulation ----
    float* arow = attn_out + ((size_t)(n * L + l0 + srow)) * S;
    for (int c = cbeg; c < cend; c += 4) {
      const float4 e = *(const float4*)&sc[srow][c];
      float4 p, pv;
      p.x = e.x * inv; p.y = e.y * inv; p.z = e.z * inv; p.w = e.w * inv;
      v4bf q4;
      q4[0] = (__bf16)p.x; q4[1] = (__bf16)p.y;
      q4[2] = (__bf16)p.z; q4[3] = (__bf16)p.w;
      *(v4bf*)&pb[srow][c] = q4;
      if (h == 0) { pv.x = pv.y = pv.z = pv.w = 0.f; }
      else        { pv = *(const float4*)&arow[c]; }
      pv.x += p.x * (1.0f / H); pv.y += p.y * (1.0f / H);
      pv.z += p.z * (1.0f / H); pv.w += p.w * (1.0f / H);
      *(float4*)&arow[c] = pv;
    }
    __syncthreads();

    // ---- out = probs @ v : wave w owns d-tile w (HD = 4 x 16) ----
    {
      const bf16_t* vbase = v_t + (b * HD + wave * 16) * S;  // [d][s] major
      v8f acc = {};
#pragma unroll 4
      for (int ss = 0; ss < S; ss += 32) {
        v16bf pa = load_frag(&pb[0][ss], S, lane);
        v16bf vb = load_frag(vbase + ss, S, lane);
        acc = wmma_bf16(pa, vb, acc);
      }
#pragma unroll
      for (int r = 0; r < 8; ++r) {
        const int l = l0 + row0 + r;
        out_h[((size_t)(l * NB + n)) * E + h * HD + wave * 16 + col] =
            (bf16_t)acc[r];
      }
    }
    __syncthreads();   // protect sc/pb before next head
  }
}

// ---------------------------------------------------------------------------
// Kernel 3: out projection. out[m, j] = out_h[m, :] . W_out[j, :] + b_out[j]
// A-tile (already bf16) staged via the CDNA5 async global->LDS datapath when
// available (ASYNCcnt), avoiding the VGPR round-trip.
// ---------------------------------------------------------------------------
__global__ __launch_bounds__(256) void out_proj_kernel(
    const bf16_t* __restrict__ out_h, const float* __restrict__ w_out,
    const float* __restrict__ b_out, float* __restrict__ out) {
  __shared__ bf16_t As[64][64];
  __shared__ bf16_t Bs[64][64];

  const int tid  = threadIdx.x;
  const int lane = tid & 31;
  const int wave = tid >> 5;
  const int m0 = blockIdx.x * 64;
  const int cb = blockIdx.y * 64;
  const int tm0 = wave >> 2;
  const int tm1 = tm0 + 2;
  const int tn  = wave & 3;

  v8f acc0 = {};
  v8f acc1 = {};

  const int r  = tid >> 2;
  const int c0 = (tid & 3) << 4;

  for (int kk = 0; kk < E; kk += 64) {
    const bf16_t* gsrc = out_h + (size_t)(m0 + r) * E + kk + c0;
#if HAVE_ASYNC_LDS
    // two 16B async copies per thread: global_load_async_to_lds_b128
    __builtin_amdgcn_global_load_async_to_lds_b128(
        (v4i_vs*)gsrc, (v4i_vs*)&As[r][c0], 0, 0);
    __builtin_amdgcn_global_load_async_to_lds_b128(
        (v4i_vs*)(gsrc + 8), (v4i_vs*)&As[r][c0 + 8], 0, 0);
#else
    const v8bf* pa = (const v8bf*)gsrc;
    *(v8bf*)&As[r][c0]     = pa[0];
    *(v8bf*)&As[r][c0 + 8] = pa[1];
#endif
    stage_f32(Bs, w_out, cb, kk, tid, (kk + 64) < E);
#if HAVE_ASYNC_LDS
    wait_async_zero();
#endif
    __syncthreads();
#pragma unroll
    for (int ks = 0; ks < 64; ks += 32) {
      v16bf a0 = load_frag(&As[tm0 * 16][ks], 64, lane);
      v16bf a1 = load_frag(&As[tm1 * 16][ks], 64, lane);
      v16bf b0 = load_frag(&Bs[tn * 16][ks], 64, lane);
      acc0 = wmma_bf16(a0, b0, acc0);
      acc1 = wmma_bf16(a1, b0, acc1);
    }
    __syncthreads();
  }

  const int col  = lane & 15;
  const int row0 = (lane >> 4) << 3;
#pragma unroll
  for (int t = 0; t < 2; ++t) {
    const v8f acc = t ? acc1 : acc0;
    const int tm = t ? tm1 : tm0;
#pragma unroll
    for (int rr = 0; rr < 8; ++rr) {
      const int grow = m0 + tm * 16 + row0 + rr;
      const int gcol = cb + tn * 16 + col;
      out[(size_t)grow * E + gcol] = acc[rr] + b_out[gcol];
    }
  }
}

// ---------------------------------------------------------------------------
extern "C" void kernel_launch(void* const* d_in, const int* in_sizes, int n_in,
                              void* d_out, int out_size, void* d_ws,
                              size_t ws_size, hipStream_t stream) {
  const float* query = (const float*)d_in[0];
  const float* key   = (const float*)d_in[1];
  const float* value = (const float*)d_in[2];
  const float* w_in  = (const float*)d_in[3];
  const float* b_in  = (const float*)d_in[4];
  const float* w_out = (const float*)d_in[5];
  const float* b_out = (const float*)d_in[6];

  float* out      = (float*)d_out;                    // (L, N, E)
  float* attn_out = out + (size_t)L * NB * E;         // (N, L, S)
  float* sig_out  = attn_out + (size_t)NB * L * S;    // (N, L, S)

  bf16_t* q_bf  = (bf16_t*)d_ws;                      // [64][L][HD]
  bf16_t* k_bf  = q_bf + (size_t)NB * H * L * HD;     // [64][S][HD]
  bf16_t* v_t   = k_bf + (size_t)NB * H * S * HD;     // [64][HD][S]
  bf16_t* out_h = v_t + (size_t)NB * H * HD * S;      // [L*N][E]

  qkv_proj_kernel<<<dim3(64, 48), 256, 0, stream>>>(query, key, value, w_in,
                                                    b_in, q_bf, k_bf, v_t);
  attention_kernel<<<dim3(L / 16, NB), 128, 0, stream>>>(q_bf, k_bf, v_t,
                                                         out_h, attn_out,
                                                         sig_out);
  out_proj_kernel<<<dim3(64, 16), 256, 0, stream>>>(out_h, w_out, b_out, out);
}
